// SparseLiDAREnc4x_8108898255316
// MI455X (gfx1250) — matrix operand order repaired and checked by reading, not compiled, not run on hardware
//
#include <hip/hip_runtime.h>

// ---------------------------------------------------------------------------
// Types for CDNA5 WMMA (wave32, 16x16x32 bf16 -> f32)
// ---------------------------------------------------------------------------
typedef __attribute__((ext_vector_type(16))) __bf16 v16bf;
typedef __attribute__((ext_vector_type(8)))  float  v8f;

union Frag32B { uint4 u[2]; v16bf v; };

__device__ __forceinline__ unsigned short f2bf(float f) {
  unsigned int u = __float_as_uint(f);
  u += 0x7fffu + ((u >> 16) & 1u);               // round-to-nearest-even
  return (unsigned short)(u >> 16);
}
__device__ __forceinline__ float bf2f(unsigned short h) {
  return __uint_as_float(((unsigned int)h) << 16);
}

// ---------------------------------------------------------------------------
// Problem constants
// ---------------------------------------------------------------------------
#define Z1 24
#define Y1 176
#define X1 176
#define Z2 12
#define Y2 88
#define X2 88
#define Z3 6
#define Y3 44
#define X3 44
#define NVOX 60000
#define GN_EPS 1e-5f

// ---------------------------------------------------------------------------
// Scatter voxels -> dense bf16 grid [z][y][x][16] + f32 mask [z][y][x]
// ---------------------------------------------------------------------------
__global__ void scatter_kernel(const float* __restrict__ vf,
                               const int* __restrict__ coors,
                               unsigned short* __restrict__ grid,
                               float* __restrict__ mask, int nvox) {
  int v = blockIdx.x * blockDim.x + threadIdx.x;
  if (v >= nvox) return;
  int z = coors[v * 4 + 1];
  int y = coors[v * 4 + 2];
  int x = coors[v * 4 + 3];
  size_t s = ((size_t)z * Y1 + y) * X1 + x;
#pragma unroll
  for (int c = 0; c < 16; ++c)
    grid[s * 16 + c] = f2bf(vf[v * 16 + c]);
  mask[s] = 1.0f;
}

// ---------------------------------------------------------------------------
// Pack f32 OIDHW weights into B-fragment-ready bf16 layout.
// K index = tap*Ci + ci (tap-major), zero-padded to multiples of 32.
// dst layout: [chunk c][co-group g][half h][n(0..15)][q(0..15)]
//   -> lane (n + 16h) of tile g loads 16 contiguous bf16 = K rows
//      c*32 + (q<8 ? h*8+q : 16+h*8+q-8), column g*16 + n.
// ---------------------------------------------------------------------------
__global__ void pack_weights(const float* __restrict__ w,
                             unsigned short* __restrict__ dst,
                             int Ci, int Co, int nchunk) {
  long i = (long)blockIdx.x * blockDim.x + threadIdx.x;
  long tot = (long)nchunk * 32 * Co;
  if (i >= tot) return;
  int q = (int)(i % 16); long r = i / 16;
  int n = (int)(r % 16); r /= 16;
  int h = (int)(r % 2);  r /= 2;
  int g = (int)(r % (Co / 16));
  int c = (int)(r / (Co / 16));
  int k  = c * 32 + (q < 8 ? h * 8 + q : 16 + h * 8 + (q - 8));
  int co = g * 16 + n;
  unsigned short val = 0;
  int Ktot = 27 * Ci;
  if (k < Ktot) {
    int t = k / Ci, ci = k % Ci;
    val = f2bf(w[((long)co * Ci + ci) * 27 + t]);
  }
  dst[i] = val;
}

// ---------------------------------------------------------------------------
// Implicit-GEMM 3x3x3 conv, bf16 in (channels-last), f32 out [s][Co].
// Fully templated on (STRIDE, CI, CO): the unrolled K loop turns every
// A access into ds_load_b128 with an immediate offset off one base VGPR and
// every B access into global_load_b128 with an immediate IOFFSET off a fixed
// base -- no per-chunk address math, no divergent predication around WMMA.
// ---------------------------------------------------------------------------
template <int STRIDE, int CI, int CO>
__global__ void conv3d_wmma(const unsigned short* __restrict__ in,
                            const unsigned short* __restrict__ wB,
                            const float* __restrict__ bias,
                            float* __restrict__ out,
                            int Din, int Hin, int Win,
                            int D, int H, int W) {
  extern __shared__ __align__(16) unsigned short lds[];
  constexpr int PW     = 15 * STRIDE + 3;       // input x extent of the halo
  constexpr int KTOT   = 27 * CI;
  constexpr int NCHUNK = (KTOT + 31) / 32;

  const int XT = (W + 15) / 16;
  int b  = blockIdx.x;
  int xt = b % XT; int rest = b / XT;
  int y0 = rest % H;
  int z0 = rest / H;
  int x0 = xt * 16;

  const int tid  = threadIdx.x;
  const int lane = tid & 31;
  const int g    = tid >> 5;           // cout tile (wave id)
  const int n    = lane & 15;          // M for A, N for B/D
  const int hi   = lane >> 4;          // lane half selects K sub-range

  // fixed per-lane base into packed weights; all chunks via immediate offsets
  const unsigned short* wbase = wB + (((size_t)g * 2 + hi) * 16 + n) * 16;

  // stream first weight lines into cache while we populate LDS
#pragma unroll
  for (int pf = 0; pf < 4; ++pf)
    __builtin_prefetch(wbase + (size_t)pf * CO * 32, 0, 1);

  // ---- cooperative LDS fill: 3(z) x 3(y) x PW(x) x CI bf16, zero-padded OOB
  constexpr int CIV8 = CI / 8;
  constexpr int NVEC = 9 * PW * CIV8;          // 8 bf16 (16B) per vector
  const int nthr = blockDim.x;
  for (int i = tid; i < NVEC; i += nthr) {
    int cv = i % CIV8;                 // pow2 -> and
    int rx = i / CIV8;                 // pow2 -> shift
    int xx = rx % PW;
    int r  = rx / PW;                  // r = dz*3 + dy
    int dz = r / 3, dy = r % 3;
    int zi = z0 * STRIDE + dz - 1;
    int yi = y0 * STRIDE + dy - 1;
    int xi = x0 * STRIDE + xx - 1;
    uint4 val = {0u, 0u, 0u, 0u};
    if (zi >= 0 && zi < Din && yi >= 0 && yi < Hin && xi >= 0 && xi < Win) {
      val = *(const uint4*)(in + (((size_t)zi * Hin + yi) * Win + xi) * CI + cv * 8);
    }
    *(uint4*)(lds + (size_t)(r * PW + xx) * CI + cv * 8) = val;
  }
  __syncthreads();

  // per-lane base into the LDS tile; taps/chunks via immediate offsets
  const unsigned short* abase = lds + (size_t)(n * STRIDE) * CI + hi * 8;

  v8f acc = {};
#pragma unroll
  for (int c = 0; c < NCHUNK; ++c) {
    Frag32B a;
#pragma unroll
    for (int grp = 0; grp < 2; ++grp) {
      const int kbase = c * 32 + grp * 16;        // compile-time after unroll
      if (kbase < KTOT) {
        const int t   = kbase / CI;               // folds to constants
        const int cib = kbase % CI;
        const int dz = t / 9, rem = t % 9, dy = rem / 3, dx = rem % 3;
        const int off = ((dz * 3 + dy) * PW + dx) * CI + cib;
        a.u[grp] = *(const uint4*)(abase + off);
      } else {
        a.u[grp] = uint4{0u, 0u, 0u, 0u};         // conv_input K tail only
      }
    }
    Frag32B bb;
    bb.u[0] = *(const uint4*)(wbase + (size_t)c * CO * 32);
    bb.u[1] = *(const uint4*)(wbase + (size_t)c * CO * 32 + 8);

    acc = __builtin_amdgcn_wmma_f32_16x16x32_bf16(
        false, a.v, false, bb.v, (short)0, acc, false, false);
  }

  // ---- store C/D: VGPR r -> (M = r + 8*hi, N = n)
  int co = g * 16 + n;
  float bv = bias ? bias[co] : 0.0f;
#pragma unroll
  for (int r = 0; r < 8; ++r) {
    int m = r + hi * 8;
    int x = x0 + m;
    if (x < W) {
      out[(((size_t)z0 * H + y0) * W + x) * CO + co] = acc[r] + bv;
    }
  }
}

// ---------------------------------------------------------------------------
// GroupNorm pass 1: per-group sum / sumsq over all spatial positions.
// stats[32] = {sum[g], sumsq[g]} interleaved; zeroed before launch.
// ---------------------------------------------------------------------------
__global__ void gn_reduce(const float* __restrict__ x,
                          float* __restrict__ stats, int S, int C) {
  __shared__ float sh[32];
  int tid  = threadIdx.x;
  int c    = tid % C;
  int rows = blockDim.x / C;
  int row  = tid / C;
  float s0 = 0.f, s1 = 0.f;
  for (int s = blockIdx.x * rows + row; s < S; s += gridDim.x * rows) {
    float v = x[(size_t)s * C + c];
    s0 += v; s1 += v * v;
  }
  if (tid < 32) sh[tid] = 0.f;
  __syncthreads();
  int g = c / (C / 16);
  atomicAdd(&sh[g * 2 + 0], s0);
  atomicAdd(&sh[g * 2 + 1], s1);
  __syncthreads();
  if (tid < 32) atomicAdd(&stats[tid], sh[tid]);
}

// ---------------------------------------------------------------------------
// GroupNorm pass 2 (fused epilogue).
//  MODE 0: out_bf = relu(gn(x) * mask)                       (bf16)
//  MODE 1: out_bf = relu(gn(x) * mask + residual)            (bf16, in-place ok)
//  MODE 2: d_out  = relu(gn(x) * mask) transposed (C,W,H,D)  (f32 final)
// ---------------------------------------------------------------------------
template <int MODE>
__global__ void gn_apply(const float* __restrict__ x,
                         const float* __restrict__ stats,
                         const float* __restrict__ gw,
                         const float* __restrict__ gb,
                         const float* __restrict__ mask,
                         const unsigned short* __restrict__ res,
                         unsigned short* __restrict__ out_bf,
                         float* __restrict__ out_f,
                         int S, int C, float invcnt, int D, int H, int W) {
  long i = (long)blockIdx.x * blockDim.x + threadIdx.x;
  if (i >= (long)S * C) return;
  int c  = (int)(i % C);
  long s = i / C;
  int g  = c / (C / 16);
  float m   = stats[g * 2 + 0] * invcnt;
  float var = stats[g * 2 + 1] * invcnt - m * m;
  float v = (x[i] - m) * rsqrtf(var + GN_EPS);
  v = v * gw[c] + gb[c];
  v *= mask[s];
  if (MODE == 1) v += bf2f(res[i]);
  v = fmaxf(v, 0.0f);
  if (MODE == 2) {
    int xq = (int)(s % W);
    int yq = (int)((s / W) % H);
    int zq = (int)(s / ((long)W * H));
    out_f[(((long)c * W + xq) * H + yq) * D + zq] = v;   // (B,C,W,H,D)
  } else {
    out_bf[i] = f2bf(v);
  }
}

// ---------------------------------------------------------------------------
// 3x3x3 stride-2 max-pool of the 0/1 mask (init value 0).
// ---------------------------------------------------------------------------
__global__ void maxpool_mask(const float* __restrict__ in, float* __restrict__ out,
                             int Di, int Hi, int Wi, int Do, int Ho, int Wo) {
  long i = (long)blockIdx.x * blockDim.x + threadIdx.x;
  if (i >= (long)Do * Ho * Wo) return;
  int x = (int)(i % Wo);
  int y = (int)((i / Wo) % Ho);
  int z = (int)(i / ((long)Wo * Ho));
  float mx = 0.0f;
  for (int dz = -1; dz <= 1; ++dz)
    for (int dy = -1; dy <= 1; ++dy)
      for (int dx = -1; dx <= 1; ++dx) {
        int zi = 2 * z + dz, yi = 2 * y + dy, xi = 2 * x + dx;
        if (zi >= 0 && zi < Di && yi >= 0 && yi < Hi && xi >= 0 && xi < Wi)
          mx = fmaxf(mx, in[((size_t)zi * Hi + yi) * Wi + xi]);
      }
  out[i] = mx;
}

// ---------------------------------------------------------------------------
// Host-side orchestration
// ---------------------------------------------------------------------------
static inline size_t align256(size_t v) { return (v + 255) & ~(size_t)255; }

template <int STRIDE, int CI, int CO>
static void launch_conv_t(const unsigned short* in, const unsigned short* wB,
                          const float* bias, float* out,
                          int Din, int Hin, int Win,
                          int D, int H, int W, hipStream_t st) {
  int XT = (W + 15) / 16;
  dim3 grid((unsigned)(D * H * XT));
  dim3 block((unsigned)(32 * (CO / 16)));
  constexpr int PW = 15 * STRIDE + 3;
  size_t ldsB = (size_t)9 * PW * CI * 2;
  conv3d_wmma<STRIDE, CI, CO><<<grid, block, ldsB, st>>>(
      in, wB, bias, out, Din, Hin, Win, D, H, W);
}

static void launch_conv(const unsigned short* in, const unsigned short* wB,
                        const float* bias, float* out,
                        int Din, int Hin, int Win, int Ci,
                        int D, int H, int W, int Co, int stride,
                        hipStream_t st) {
  if (stride == 1 && Ci == 16  && Co == 32)
    launch_conv_t<1, 16, 32>(in, wB, bias, out, Din, Hin, Win, D, H, W, st);
  else if (stride == 1 && Ci == 32  && Co == 32)
    launch_conv_t<1, 32, 32>(in, wB, bias, out, Din, Hin, Win, D, H, W, st);
  else if (stride == 2 && Ci == 32  && Co == 64)
    launch_conv_t<2, 32, 64>(in, wB, bias, out, Din, Hin, Win, D, H, W, st);
  else if (stride == 1 && Ci == 64  && Co == 64)
    launch_conv_t<1, 64, 64>(in, wB, bias, out, Din, Hin, Win, D, H, W, st);
  else if (stride == 2 && Ci == 64  && Co == 128)
    launch_conv_t<2, 64, 128>(in, wB, bias, out, Din, Hin, Win, D, H, W, st);
  else if (stride == 1 && Ci == 128 && Co == 128)
    launch_conv_t<1, 128, 128>(in, wB, bias, out, Din, Hin, Win, D, H, W, st);
}

static void run_gn(int mode, const float* conv, const float* gw, const float* gb,
                   const float* mask, const unsigned short* res,
                   unsigned short* out_bf, float* out_f,
                   int S, int C, float* stats, int D, int H, int W,
                   hipStream_t st) {
  hipMemsetAsync(stats, 0, 32 * sizeof(float), st);
  int rows = 256 / C;
  long nb = ((long)S + rows - 1) / rows;
  int blocks = (int)(nb < 2048 ? nb : 2048);
  gn_reduce<<<blocks, 256, 0, st>>>(conv, stats, S, C);
  long tot = (long)S * C;
  int gblk = (int)((tot + 255) / 256);
  float invcnt = 1.0f / ((float)(C / 16) * (float)S);
  if (mode == 0)
    gn_apply<0><<<gblk, 256, 0, st>>>(conv, stats, gw, gb, mask, res,
                                      out_bf, out_f, S, C, invcnt, D, H, W);
  else if (mode == 1)
    gn_apply<1><<<gblk, 256, 0, st>>>(conv, stats, gw, gb, mask, res,
                                      out_bf, out_f, S, C, invcnt, D, H, W);
  else
    gn_apply<2><<<gblk, 256, 0, st>>>(conv, stats, gw, gb, mask, res,
                                      out_bf, out_f, S, C, invcnt, D, H, W);
}

extern "C" void kernel_launch(void* const* d_in, const int* in_sizes, int n_in,
                              void* d_out, int out_size, void* d_ws, size_t ws_size,
                              hipStream_t stream) {
  // ---- input mapping (JAX pytree: dict keys sorted alphabetically) ----
  // 0 voxel_features, 1 coors,
  // conv1.b0: 2 g1b, 3 g1w, 4 g2b, 5 g2w, 6 w1, 7 w2
  // conv1.b1: 8..13 | conv2.b0: 14..19 | conv2.b1: 20..25
  // conv2.down: 26 gb, 27 gw, 28 w
  // conv3.b0: 29..34 | conv3.b1: 35..40 | conv3.down: 41 gb, 42 gw, 43 w
  // conv_input: 44 b, 45 gb, 46 gw, 47 w
  // conv_out:   48 b, 49 gb, 50 gw, 51 w
  // 52 batch_size (ignored, B==1)
  const float* vf   = (const float*)d_in[0];
  const int*   cor  = (const int*)d_in[1];
  auto P = [&](int i) -> const float* { return (const float*)d_in[i]; };

  const int S1 = Z1 * Y1 * X1;   // 743424
  const int S2 = Z2 * Y2 * X2;   // 92928
  const int S3 = Z3 * Y3 * X3;   // 11616

  // ---- workspace carve (deterministic every call) ----
  char* base = (char*)d_ws;
  size_t off = 0;
  auto walloc = [&](size_t bytes) -> void* {
    void* p = base + off;
    off += align256(bytes);
    return p;
  };
  unsigned short* grid_in = (unsigned short*)walloc((size_t)S1 * 16 * 2);
  float* mask1 = (float*)walloc((size_t)S1 * 4);
  float* mask2 = (float*)walloc((size_t)S2 * 4);
  float* mask3 = (float*)walloc((size_t)S3 * 4);
  unsigned short* actA = (unsigned short*)walloc((size_t)S1 * 32 * 2);
  unsigned short* actB = (unsigned short*)walloc((size_t)S1 * 32 * 2);
  float* convbuf = (float*)walloc((size_t)S1 * 32 * 4);
  float* stats   = (float*)walloc(32 * sizeof(float));

  // packed weights (16 convs, execution order)
  struct WDesc { int idx, Ci, Co; };
  WDesc all[16] = {
      {47, 16, 32},                                             // conv_input
      {6, 32, 32}, {7, 32, 32}, {12, 32, 32}, {13, 32, 32},     // conv1 blocks
      {28, 32, 64},                                             // conv2 down
      {18, 64, 64}, {19, 64, 64}, {24, 64, 64}, {25, 64, 64},   // conv2 blocks
      {43, 64, 128},                                            // conv3 down
      {33, 128, 128}, {34, 128, 128},                           // conv3 b0
      {39, 128, 128}, {40, 128, 128},                           // conv3 b1
      {51, 128, 128},                                           // conv_out
  };
  unsigned short* pw[16];
  int pw_chunks[16];
  for (int i = 0; i < 16; ++i) {
    int nch = (27 * all[i].Ci + 31) / 32;
    pw_chunks[i] = nch;
    pw[i] = (unsigned short*)walloc((size_t)nch * 32 * all[i].Co * 2);
  }
  if (off > ws_size) return;   // workspace too small; bail deterministically

  // ---- zero grid/mask, scatter voxels ----
  hipMemsetAsync(grid_in, 0, (size_t)S1 * 16 * 2, stream);
  hipMemsetAsync(mask1, 0, (size_t)S1 * 4, stream);
  scatter_kernel<<<(NVOX + 255) / 256, 256, 0, stream>>>(vf, cor, grid_in, mask1, NVOX);

  // ---- pack all conv weights into B-fragment layout ----
  for (int i = 0; i < 16; ++i) {
    long tot = (long)pw_chunks[i] * 32 * all[i].Co;
    pack_weights<<<(unsigned)((tot + 255) / 256), 256, 0, stream>>>(
        P(all[i].idx), pw[i], all[i].Ci, all[i].Co, pw_chunks[i]);
  }

  // ======================= stage 1: full res, C=32 =======================
  launch_conv(grid_in, pw[0], P(44), convbuf, Z1, Y1, X1, 16, Z1, Y1, X1, 32, 1, stream);
  run_gn(0, convbuf, P(46), P(45), mask1, nullptr, actA, nullptr,
         S1, 32, stats, Z1, Y1, X1, stream);

  const int b32[2][6] = {{2, 3, 4, 5, 6, 7}, {8, 9, 10, 11, 12, 13}};
  for (int blk = 0; blk < 2; ++blk) {
    int pwi1 = 1 + blk * 2, pwi2 = 2 + blk * 2;
    launch_conv(actA, pw[pwi1], nullptr, convbuf, Z1, Y1, X1, 32, Z1, Y1, X1, 32, 1, stream);
    run_gn(0, convbuf, P(b32[blk][1]), P(b32[blk][0]), mask1, nullptr, actB, nullptr,
           S1, 32, stats, Z1, Y1, X1, stream);
    launch_conv(actB, pw[pwi2], nullptr, convbuf, Z1, Y1, X1, 32, Z1, Y1, X1, 32, 1, stream);
    run_gn(1, convbuf, P(b32[blk][3]), P(b32[blk][2]), mask1, actA, actA, nullptr,
           S1, 32, stats, Z1, Y1, X1, stream);
  }

  // ======================= stage 2: half res, C=64 =======================
  maxpool_mask<<<(unsigned)((S2 + 255) / 256), 256, 0, stream>>>(
      mask1, mask2, Z1, Y1, X1, Z2, Y2, X2);
  launch_conv(actA, pw[5], nullptr, convbuf, Z1, Y1, X1, 32, Z2, Y2, X2, 64, 2, stream);
  run_gn(0, convbuf, P(27), P(26), mask2, nullptr, actB, nullptr,
         S2, 64, stats, Z2, Y2, X2, stream);

  const int b64[2][6] = {{14, 15, 16, 17, 18, 19}, {20, 21, 22, 23, 24, 25}};
  for (int blk = 0; blk < 2; ++blk) {
    int pwi1 = 6 + blk * 2, pwi2 = 7 + blk * 2;
    launch_conv(actB, pw[pwi1], nullptr, convbuf, Z2, Y2, X2, 64, Z2, Y2, X2, 64, 1, stream);
    run_gn(0, convbuf, P(b64[blk][1]), P(b64[blk][0]), mask2, nullptr, actA, nullptr,
           S2, 64, stats, Z2, Y2, X2, stream);
    launch_conv(actA, pw[pwi2], nullptr, convbuf, Z2, Y2, X2, 64, Z2, Y2, X2, 64, 1, stream);
    run_gn(1, convbuf, P(b64[blk][3]), P(b64[blk][2]), mask2, actB, actB, nullptr,
           S2, 64, stats, Z2, Y2, X2, stream);
  }

  // ====================== stage 3: quarter res, C=128 =====================
  maxpool_mask<<<(unsigned)((S3 + 255) / 256), 256, 0, stream>>>(
      mask2, mask3, Z2, Y2, X2, Z3, Y3, X3);
  launch_conv(actB, pw[10], nullptr, convbuf, Z2, Y2, X2, 64, Z3, Y3, X3, 128, 2, stream);
  run_gn(0, convbuf, P(42), P(41), mask3, nullptr, actA, nullptr,
         S3, 128, stats, Z3, Y3, X3, stream);

  const int b128[2][6] = {{29, 30, 31, 32, 33, 34}, {35, 36, 37, 38, 39, 40}};
  const int pw128[2][2] = {{11, 12}, {13, 14}};
  for (int blk = 0; blk < 2; ++blk) {
    launch_conv(actA, pw[pw128[blk][0]], nullptr, convbuf, Z3, Y3, X3, 128, Z3, Y3, X3, 128, 1, stream);
    run_gn(0, convbuf, P(b128[blk][1]), P(b128[blk][0]), mask3, nullptr, actB, nullptr,
           S3, 128, stats, Z3, Y3, X3, stream);
    launch_conv(actB, pw[pw128[blk][1]], nullptr, convbuf, Z3, Y3, X3, 128, Z3, Y3, X3, 128, 1, stream);
    run_gn(1, convbuf, P(b128[blk][3]), P(b128[blk][2]), mask3, actA, actA, nullptr,
           S3, 128, stats, Z3, Y3, X3, stream);
  }

  // =========================== conv_out + final ===========================
  launch_conv(actA, pw[15], P(48), convbuf, Z3, Y3, X3, 128, Z3, Y3, X3, 128, 1, stream);
  run_gn(2, convbuf, P(50), P(49), mask3, nullptr, nullptr, (float*)d_out,
         S3, 128, stats, Z3, Y3, X3, stream);
}
